// RNN_55937654063621
// MI455X (gfx1250) — matrix operand order repaired and checked by reading, not compiled, not run on hardware
//
#include <hip/hip_runtime.h>
#include <hip/hip_bf16.h>
#include <math.h>

#define BATCH  512
#define SEQ    512
#define EMBED  32
#define HIDDEN 128

typedef __attribute__((ext_vector_type(16))) __bf16 v16bf;
typedef __attribute__((ext_vector_type(2)))  __bf16 bf16x2;
typedef __attribute__((ext_vector_type(8)))  float  v8f;
typedef __attribute__((ext_vector_type(2)))  float  f32x2;

// 16-bit A-matrix 16x32 layout (ISA 7.12.2): lane m=lane&15; lane-group g adds K+8;
// VGPR j holds K-pair {k0,k0+1}, k0 = 2j (j<4) else 8+2j.
__device__ __forceinline__ int kmapA(int j, int g) {
  return 2 * j + ((j >= 4) ? 8 : 0) + 8 * g;
}
// 16-bit B-matrix 32x16 layout: lane n=lane&15; lane-group g adds K+8;
// VGPR j holds K-pair {k0,k0+1}, k0 = 2j (j<4) else 8+2j (same as A).
__device__ __forceinline__ int kmapB(int j, int g) {
  return 2 * j + ((j >= 4) ? 8 : 0) + 8 * g;
}

// Branch-free tanh. Prefer the gfx1250 hardware transcendental v_tanh_f32;
// otherwise tanh(x) = 1 - 2/(exp(2x)+1) via v_exp_f32 + v_rcp_f32, which is
// monotonic and saturates to +-1 for large |x| with no EXEC-divergent paths.
__device__ __forceinline__ float fast_tanh(float v) {
#if __has_builtin(__builtin_amdgcn_tanhf)
  return __builtin_amdgcn_tanhf(v);
#else
  float e = __expf(2.0f * v);
  return 1.0f - 2.0f / (e + 1.0f);
#endif
}

__global__ __launch_bounds__(256) void rnn_elman_wmma(
    const float* __restrict__ x,   // [B, T, E]
    const float* __restrict__ Wx,  // [E, H]
    const float* __restrict__ Wh,  // [H, H]
    const float* __restrict__ bh,  // [H]
    float* __restrict__ out,       // [B, T, H]
    float* __restrict__ hfinal)    // [B, H]
{
  __shared__ __align__(16) __bf16 hb[16 * HIDDEN];    // h_{t-1}, bf16, row-major
  __shared__ __align__(16) __bf16 xb[2][16 * EMBED];  // double-buffered x_t tile

  const int tid  = threadIdx.x;
  const int lane = tid & 31;
  const int wave = tid >> 5;        // 0..7 -> owns N columns [wave*16, wave*16+16)
  const int lm   = lane & 15;       // C/D column within tile; A row
  const int grp  = lane >> 4;       // lane group
  const int n0   = wave * 16;
  const int b0   = blockIdx.x * 16; // batch tile base

  // ---- one-time: W_h (4 K-chunks) and W_x B-fragments -> bf16 in VGPRs ----
  v16bf bWh[4];
#pragma unroll
  for (int c = 0; c < 4; ++c) {
#pragma unroll
    for (int j = 0; j < 8; ++j) {
      int k = 32 * c + kmapB(j, grp);
      bWh[c][2 * j]     = (__bf16)Wh[(size_t)(k)     * HIDDEN + n0 + lm];
      bWh[c][2 * j + 1] = (__bf16)Wh[(size_t)(k + 1) * HIDDEN + n0 + lm];
    }
  }
  v16bf bWx;
#pragma unroll
  for (int j = 0; j < 8; ++j) {
    int k = kmapB(j, grp);
    bWx[2 * j]     = (__bf16)Wx[(size_t)(k)     * HIDDEN + n0 + lm];
    bWx[2 * j + 1] = (__bf16)Wx[(size_t)(k + 1) * HIDDEN + n0 + lm];
  }
  const float bias = bh[n0 + lm];

  // ---- h0 = 0; stage x_0 tile to LDS (bf16) ----
  for (int i = tid; i < 16 * HIDDEN; i += 256) hb[i] = (__bf16)0.0f;
  {
    int i = tid * 2;                // 16*32 = 512 elems, 2 consecutive per thread
    int m = i >> 5, e = i & 31;
    f32x2 v = *reinterpret_cast<const f32x2*>(
        x + (size_t)(b0 + m) * SEQ * EMBED + e);
    xb[0][i]     = (__bf16)v.x;
    xb[0][i + 1] = (__bf16)v.y;
  }
  __syncthreads();

  for (int t = 0; t < SEQ; ++t) {
    const __bf16* xcur = xb[t & 1];

    // acc = bias broadcast over the 8 row-VGPRs of the C/D tile
    v8f acc;
#pragma unroll
    for (int r = 0; r < 8; ++r) acc[r] = bias;

    // input projection: A = x_t tile [16x32] (K = E = 32 exactly)
    v16bf a;
#pragma unroll
    for (int j = 0; j < 8; ++j) {
      int k = kmapA(j, grp);
      bf16x2 p = *reinterpret_cast<const bf16x2*>(xcur + lm * EMBED + k);
      a[2 * j] = p.x; a[2 * j + 1] = p.y;
    }
    acc = __builtin_amdgcn_wmma_f32_16x16x32_bf16(false, a, false, bWx,
                                                  (short)0, acc, false, false);

    // recurrent projection: K = 128 as 4 chunks of 32
#pragma unroll
    for (int c = 0; c < 4; ++c) {
#pragma unroll
      for (int j = 0; j < 8; ++j) {
        int k = 32 * c + kmapA(j, grp);
        bf16x2 p = *reinterpret_cast<const bf16x2*>(hb + lm * HIDDEN + k);
        a[2 * j] = p.x; a[2 * j + 1] = p.y;
      }
      acc = __builtin_amdgcn_wmma_f32_16x16x32_bf16(false, a, false, bWh[c],
                                                    (short)0, acc, false, false);
    }

    // prefetch x_{t+1} into registers (overlaps WMMA drain / tanh)
    f32x2 xnext = {0.0f, 0.0f};
    if (t + 1 < SEQ) {
      int i = tid * 2, m = i >> 5, e = i & 31;
      xnext = *reinterpret_cast<const f32x2*>(
          x + (size_t)(b0 + m) * SEQ * EMBED + (size_t)(t + 1) * EMBED + e);
    }

    // activation + stream h_t to global (this is the HBM-bound 134 MB stream)
#pragma unroll
    for (int r = 0; r < 8; ++r) {
      float h = fast_tanh(acc[r]);
      acc[r] = h;
      int m = r + 8 * grp;
      out[((size_t)(b0 + m) * SEQ + t) * HIDDEN + n0 + lm] = h;
    }

    if (t == SEQ - 1) {
      // final hidden state in full f32 precision straight from the accumulator
#pragma unroll
      for (int r = 0; r < 8; ++r) {
        int m = r + 8 * grp;
        hfinal[(size_t)(b0 + m) * HIDDEN + n0 + lm] = acc[r];
      }
    }

    __syncthreads();  // all waves done reading h_{t-1} / x_t from LDS
#pragma unroll
    for (int r = 0; r < 8; ++r) {
      int m = r + 8 * grp;
      hb[m * HIDDEN + n0 + lm] = (__bf16)acc[r];
    }
    if (t + 1 < SEQ) {
      int i = tid * 2;
      xb[(t + 1) & 1][i]     = (__bf16)xnext.x;
      xb[(t + 1) & 1][i + 1] = (__bf16)xnext.y;
    }
    __syncthreads();  // h_t and x_{t+1} visible to all 8 waves
  }
}

extern "C" void kernel_launch(void* const* d_in, const int* in_sizes, int n_in,
                              void* d_out, int out_size, void* d_ws, size_t ws_size,
                              hipStream_t stream) {
  const float* x  = (const float*)d_in[0];
  const float* Wx = (const float*)d_in[1];
  const float* Wh = (const float*)d_in[2];
  const float* bh = (const float*)d_in[3];
  float* out  = (float*)d_out;
  float* hfin = out + (size_t)BATCH * SEQ * HIDDEN;

  dim3 grid(BATCH / 16);   // 32 workgroups, one 16-row batch tile each
  dim3 block(256);         // 8 wave32s; wave w owns hidden columns [16w,16w+16)
  hipLaunchKernelGGL(rnn_elman_wmma, grid, block, 0, stream,
                     x, Wx, Wh, bh, out, hfin);
}
